// TGraphGen_910533067546
// MI455X (gfx1250) — matrix operand order repaired and checked by reading, not compiled, not run on hardware
//
#include <hip/hip_runtime.h>
#include <hip/hip_bf16.h>

typedef __attribute__((ext_vector_type(16))) _Float16 v16h;
typedef __attribute__((ext_vector_type(8)))  _Float16 v8h;
typedef __attribute__((ext_vector_type(8)))  float    v8f;
typedef __attribute__((ext_vector_type(4)))  float    v4f;

typedef unsigned int u32x4 __attribute__((ext_vector_type(4)));
typedef int          i32x4 __attribute__((ext_vector_type(4)));
typedef int          i32x8 __attribute__((ext_vector_type(8)));

#define N_   64
#define C_   256
#define T_   64
#define V_   25
#define CS_  64
#define M_   (T_ * V_)          // 1600 columns of X_n
#define YMAT ((size_t)(N_ * V_) * T_ * CS_)  // halves per Y matrix = 1600*4096

__device__ __forceinline__ v16h cat8(v8h lo, v8h hi) {
    return __builtin_shufflevector(lo, hi,
        0,1,2,3,4,5,6,7,8,9,10,11,12,13,14,15);
}

// ---- TDM: 1-D async copy of nelem f16 from global to LDS (ISA ch.8 D#) ----
__device__ __forceinline__ void tdm_load_f16_1d(const _Float16* gsrc,
                                                unsigned lds_byte_off,
                                                unsigned nelem)
{
    unsigned long long ga = (unsigned long long)(uintptr_t)gsrc;
    // D# group 0: count=1 (valid, user mode), lds_addr, global_addr, type=2
    u32x4 g0 = { 1u,
                 lds_byte_off,
                 (unsigned)ga,
                 (unsigned)((ga >> 32) & 0x01FFFFFFu) | (2u << 30) };
    // D# group 1: data_size=1 (2B), tensor_dim0=nelem, tensor_dim1=1,
    //             tile_dim0=nelem (1-D tile), tensor_dim0_stride=nelem
    i32x8 g1 = { (int)(1u << 16),
                 (int)(nelem << 16),          // tensor_dim0[15:0] @ bits 63:48
                 (int)(1u << 16),             // tensor_dim0[31:16]=0, tensor_dim1=1
                 (int)(nelem << 16),          // tile_dim0 @ bits 127:112
                 0,                           // tile_dim1=0, tile_dim2=0 (unused)
                 (int)nelem,                  // tensor_dim0_stride[31:0]
                 0, 0 };
    i32x4 gz = { 0, 0, 0, 0 };
#if __clang_major__ >= 23
    i32x8 gz8 = { 0, 0, 0, 0, 0, 0, 0, 0 };
    __builtin_amdgcn_tensor_load_to_lds(g0, g1, gz, gz, gz8, 0);
#else
    __builtin_amdgcn_tensor_load_to_lds(g0, g1, gz, gz, 0);
#endif
}

// ---------------- prep: f32 -> f16 weights into workspace ----------------
__global__ __launch_bounds__(256) void wconv_kernel(
    const float* __restrict__ Wphi, const float* __restrict__ Wpsi,
    _Float16* __restrict__ Wf)  // [2][CS][C]
{
    int i = blockIdx.x * 256 + threadIdx.x;
    if (i < CS_ * C_) {
        Wf[i]            = (_Float16)Wphi[i];
        Wf[CS_ * C_ + i] = (_Float16)Wpsi[i];
    }
}

// ---------------- stage 1: Y{phi,psi} = W @ X_n, stored [nv][t][o] f16 ----
__global__ __launch_bounds__(256) void proj_kernel(
    const float* __restrict__ x,       // [N][C][T][V] = [N][C][M]
    const _Float16* __restrict__ Wf,   // [2][CS][C]
    _Float16* __restrict__ Y)          // [2][N*V][T][CS]
{
    __shared__ _Float16 Xs[64][264];   // [m][c], padded rows (528B, 16B aligned)

    const int mtile = blockIdx.x;      // 0..24
    const int n     = blockIdx.y;      // 0..63
    const int m0    = mtile * 64;
    const int tid   = threadIdx.x;
    const int lane  = tid & 31;
    const int w     = tid >> 5;
    const int l15   = lane & 15;
    const int lh    = (lane < 16) ? 0 : 1;

    // ---- load 256x64 f32 tile of X_n, coalesced, transpose to f16 LDS ----
    {
        const float* xn = x + (size_t)n * C_ * M_ + m0;
        const int j4   = (tid & 15) * 4;   // m within tile
        const int crow = tid >> 4;         // c row group
        for (int p = 0; p < 16; ++p) {
            int c = crow + p * 16;
            v4f xv = *(const v4f*)(xn + (size_t)c * M_ + j4);
            Xs[j4 + 0][c] = (_Float16)xv.x;
            Xs[j4 + 1][c] = (_Float16)xv.y;
            Xs[j4 + 2][c] = (_Float16)xv.z;
            Xs[j4 + 3][c] = (_Float16)xv.w;
        }
    }
    __syncthreads();

    // ---- 32 output tiles (2 mats x 4 mo x 4 nm), 4 per wave ----
    for (int i = 0; i < 4; ++i) {
        const int tile = i * 8 + w;
        const int mat  = tile >> 4;        // 0 = phi, 1 = psi
        const int rem  = tile & 15;
        const int mo   = rem >> 2;         // o tile (A rows)
        const int nm   = rem & 3;          // m tile (B cols)

        // A: W rows, ISA 16-bit A layout (lane<16 -> K base 0, >=16 -> +8)
        const _Float16* pA = Wf + (size_t)mat * (CS_ * C_)
                              + (size_t)(mo * 16 + l15) * C_ + lh * 8;
        // B: Xs rows (lane = m column), ISA 16-bit layout (lane<16 -> K base 0, >=16 -> +8)
        const _Float16* pB = &Xs[nm * 16 + l15][lh * 8];

        v8f acc = {0.f,0.f,0.f,0.f,0.f,0.f,0.f,0.f};
        for (int c0 = 0; c0 < C_; c0 += 32) {
            v8h alo = *(const v8h*)(pA + c0);
            v8h ahi = *(const v8h*)(pA + c0 + 16);
            v8h blo = *(const v8h*)(pB + c0);
            v8h bhi = *(const v8h*)(pB + c0 + 16);
            v16h a = cat8(alo, ahi);
            v16h b = cat8(blo, bhi);
            acc = __builtin_amdgcn_wmma_f32_16x16x32_f16(
                false, a, false, b, (short)0, acc, false, false);
        }

        // D layout: lane owns 8 consecutive o rows -> packed 16B store,
        // giving the [t][o] transpose for free.
        const int mg = m0 + nm * 16 + l15;   // global column m = t*25 + v
        const int t  = mg / V_;
        const int v  = mg % V_;
        v8h hv;
        for (int r = 0; r < 8; ++r) hv[r] = (_Float16)acc[r];
        _Float16* dst = Y + (size_t)mat * YMAT
                          + ((size_t)(n * V_ + v) * T_ + t) * CS_
                          + mo * 16 + lh * 8;
        *(v8h*)dst = hv;
    }
}

// ---------------- stage 2: S = Y1^T Y2 per (n,v), row-L2-normalized -------
__global__ __launch_bounds__(256) void corr_kernel(
    const _Float16* __restrict__ Y,    // [2][N*V][T][CS]
    float* __restrict__ out)           // [N*V][T][T]
{
    __shared__ _Float16 Ys[2][64][64]; // [mat][t][o], 16 KB, filled by TDM
    __shared__ float rowSq[64];

    const int nv   = blockIdx.x;       // 0..1599
    const int tid  = threadIdx.x;
    const int lane = tid & 31;
    const int w    = tid >> 5;
    const int l15  = lane & 15;
    const int lh   = (lane < 16) ? 0 : 1;

    if (tid < 64) rowSq[tid] = 0.0f;

    // wave 0 drives the Tensor Data Mover: two 8KB contiguous tiles -> LDS
    if (w == 0) {
        tdm_load_f16_1d(Y + (size_t)nv * (T_ * CS_),
                        (unsigned)(uintptr_t)&Ys[0][0][0], T_ * CS_);
        tdm_load_f16_1d(Y + YMAT + (size_t)nv * (T_ * CS_),
                        (unsigned)(uintptr_t)&Ys[1][0][0], T_ * CS_);
        __builtin_amdgcn_s_wait_tensorcnt(0);
    }
    __syncthreads();

    v8f acc[2];
    for (int i = 0; i < 2; ++i) {
        const int tile = w + i * 8;        // 16 tiles, 2 per wave
        const int mo   = tile >> 2;        // t1 tile
        const int no   = tile & 3;         // t2 tile

        const _Float16* pA = &Ys[0][mo * 16 + l15][lh * 8];
        const _Float16* pB = &Ys[1][no * 16 + l15][lh * 8];

        v8f a_ = {0.f,0.f,0.f,0.f,0.f,0.f,0.f,0.f};
        for (int k0 = 0; k0 < CS_; k0 += 32) {
            v8h alo = *(const v8h*)(pA + k0);
            v8h ahi = *(const v8h*)(pA + k0 + 16);
            v8h blo = *(const v8h*)(pB + k0);
            v8h bhi = *(const v8h*)(pB + k0 + 16);
            v16h a = cat8(alo, ahi);
            v16h b = cat8(blo, bhi);
            a_ = __builtin_amdgcn_wmma_f32_16x16x32_f16(
                false, a, false, b, (short)0, a_, false, false);
        }
        acc[i] = a_;

        // accumulate row sums of squares (rows = t1 = mo*16 + lh*8 + r)
        for (int r = 0; r < 8; ++r) {
            float e = a_[r];
            atomicAdd(&rowSq[mo * 16 + lh * 8 + r], e * e);
        }
    }
    __syncthreads();

    for (int i = 0; i < 2; ++i) {
        const int tile = w + i * 8;
        const int mo   = tile >> 2;
        const int no   = tile & 3;
        for (int r = 0; r < 8; ++r) {
            const int row = mo * 16 + lh * 8 + r;   // t1
            const float inv = 1.0f / fmaxf(sqrtf(rowSq[row]), 1e-12f);
            out[(size_t)nv * (T_ * T_) + (size_t)row * T_ + no * 16 + l15]
                = acc[i][r] * inv;
        }
    }
}

extern "C" void kernel_launch(void* const* d_in, const int* in_sizes, int n_in,
                              void* d_out, int out_size, void* d_ws, size_t ws_size,
                              hipStream_t stream) {
    const float* x    = (const float*)d_in[0];
    const float* Wphi = (const float*)d_in[1];
    const float* Wpsi = (const float*)d_in[2];
    float* out = (float*)d_out;

    _Float16* Wf = (_Float16*)d_ws;                              // 64 KB
    _Float16* Y  = (_Float16*)((char*)d_ws + (size_t)2 * CS_ * C_ * sizeof(_Float16));
    (void)in_sizes; (void)n_in; (void)out_size; (void)ws_size;

    wconv_kernel<<<64, 256, 0, stream>>>(Wphi, Wpsi, Wf);
    proj_kernel<<<dim3(V_, N_), 256, 0, stream>>>(x, Wf, Y);
    corr_kernel<<<N_ * V_, 256, 0, stream>>>(Y, out);
}
